// GATNet_2688649527832
// MI455X (gfx1250) — matrix operand order repaired and checked by reading, not compiled, not run on hardware
//
#include <hip/hip_runtime.h>
#include <hip/hip_bf16.h>
#include <math.h>

// ---------------- types ----------------
typedef __attribute__((ext_vector_type(16))) _Float16 v16h;
typedef __attribute__((ext_vector_type(8)))  _Float16 v8h;
typedef __attribute__((ext_vector_type(8)))  float    v8f;

#define GAT_IN    128
#define GAT_HID   32
#define GAT_HEADS 8
#define GAT_F1    (GAT_HEADS * GAT_HID)   // 256
#define GAT_OUT   64
#define NEG_SLOPE 0.2f
#define GAT_EPS   1e-16f

static __host__ __device__ inline int ceil_div(int a, int b) { return (a + b - 1) / b; }

// ---------------- elementwise convert f32 -> f16 (activations) ----------------
__global__ void k_cvt_f32_to_f16(const float* __restrict__ in, _Float16* __restrict__ out, int n) {
    int i = blockIdx.x * blockDim.x + threadIdx.x;
    if (i < n) out[i] = (_Float16)in[i];
}

// ---------------- pack weights f32 [K,Nc] -> f16 WMMA-B fragment layout ----------------
// Bp layout: [K/32][Nc/16][n 0..15][k 0..31] halfs.  This code's contiguous k order (not the ISA table order), lane=column,
// lanes 0-15 take K 0..15 and lanes 16-31 take K 16..31 of each 32-K slab, so each
// lane's 16 fragment halfs are contiguous (two global_load_b128).
__global__ void k_pack_b(const float* __restrict__ B, _Float16* __restrict__ Bp, int K, int Nc) {
    int t = blockIdx.x * blockDim.x + threadIdx.x;
    if (t >= K * Nc) return;
    int k = t / Nc, n = t % Nc;
    int idx = (((k >> 5) * (Nc >> 4) + (n >> 4)) * 16 + (n & 15)) * 32 + (k & 31);
    Bp[idx] = (_Float16)B[t];
}

// ---------------- WMMA GEMM: C[M,Nc] = A[M,K] * Bpacked ----------------
// f16 inputs, f32 output.  K % 32 == 0, Nc % 64 == 0, M guarded at the store.
// Block = 256 threads (8 waves), block tile 64(M) x 64(N); each wave computes
// 16x32 (two accumulators sharing one A fragment).
// A is staged to LDS with the CDNA5 async global->LDS path
// (global_load_async_to_lds_b128, ASYNCcnt), in natural k order so each
// lane's fragment is one contiguous 32B read.  Out-of-range M rows are clamped
// (their D rows are never stored), keeping EXEC uniform for the async op.
// B fragments come straight from the packed (L2-resident) weight buffer.
__global__ __launch_bounds__(256)
void k_gemm_f16_wmma(const _Float16* __restrict__ A, const _Float16* __restrict__ Bp,
                     float* __restrict__ C, int M, int K, int Nc) {
    // per row: k stored in natural order [0..31]  -> lane frag = 16 contiguous halfs
    __shared__ __align__(64) _Float16 sA[64 * 32];

    const int m0     = blockIdx.x * 64;
    const int n0     = blockIdx.y * 64;
    const int nTiles = Nc >> 4;

    const int w     = threadIdx.x >> 5;          // wave 0..7
    const int lane  = threadIdx.x & 31;
    const int mTile = w & 3;                     // 0..3
    const int nBase = (n0 >> 4) + (w >> 2) * 2;  // global 16-col tile index (2 per wave)
    const int mrow  = lane & 15;
    const int hi    = lane >> 4;                 // half-wave selector
    const int ncol  = lane & 15;

    // A cooperative staging: 64 rows x 4 chunks of 8 halfs, row clamped to M-1
    const int aRow   = threadIdx.x >> 2;
    const int aChunk = threadIdx.x & 3;
    const int aDst   = aChunk * 8;                                  // 0,8,16,24
    int gA = m0 + aRow; if (gA >= M) gA = M - 1;
    const unsigned ldsOff = (unsigned)(size_t)&sA[aRow * 32 + aDst];

    v8f acc0 = {}, acc1 = {};

    for (int k0 = 0; k0 < K; k0 += 32) {
        __syncthreads();                         // previous iter fragment reads done
        {   // async DMA: 16B per lane, global -> LDS, tracked by ASYNCcnt
            const _Float16* gptr = &A[(size_t)gA * K + k0 + aChunk * 8];
            asm volatile("global_load_async_to_lds_b128 %0, %1, off"
                         :: "v"(ldsOff), "v"(gptr) : "memory");
        }
        if (k0 + 32 < K)                         // emits global_prefetch_b8
            __builtin_prefetch(&A[(size_t)gA * K + k0 + 32 + aChunk * 8]);
        asm volatile("s_wait_asynccnt 0x0" ::: "memory");
        __syncthreads();

        // A fragment: 32B contiguous -> 2x ds_load_b128
        v16h af = *(const v16h*)&sA[(mTile * 16 + mrow) * 32 + hi * 16];
        // B fragments: 32B contiguous each -> 2x global_load_b128 (L2-hot weights)
        const size_t kt = (size_t)(k0 >> 5) * nTiles;
        v16h bf0 = *(const v16h*)&Bp[((kt + nBase    ) * 16 + ncol) * 32 + hi * 16];
        v16h bf1 = *(const v16h*)&Bp[((kt + nBase + 1) * 16 + ncol) * 32 + hi * 16];

        acc0 = __builtin_amdgcn_wmma_f32_16x16x32_f16(false, af, false, bf0,
                                                      (short)0, acc0, false, false);
        acc1 = __builtin_amdgcn_wmma_f32_16x16x32_f16(false, af, false, bf1,
                                                      (short)0, acc1, false, false);
    }

    // D layout: VGPR r -> row (hi?8:0)+r, col = lane&15
    const int rowBase = m0 + mTile * 16 + hi * 8;
    const int col0    = nBase * 16 + ncol;
#pragma unroll
    for (int r = 0; r < 8; ++r) {
        int gr = rowBase + r;
        if (gr < M) {
            C[(size_t)gr * Nc + col0]      = acc0[r];
            C[(size_t)gr * Nc + col0 + 16] = acc1[r];
        }
    }
}

// ---------------- per-(node,head) attention logits ----------------
__global__ void k_attn_logits(const float* __restrict__ xw,
                              const float* __restrict__ a_src,
                              const float* __restrict__ a_dst,
                              float* __restrict__ alS, float* __restrict__ alD,
                              int N, int H, int C) {
    int t = blockIdx.x * blockDim.x + threadIdx.x;
    if (t >= N * H) return;
    int n = t / H, h = t % H;
    const float* xv = &xw[(size_t)n * H * C + h * C];
    const float* as = &a_src[h * C];
    const float* ad = &a_dst[h * C];
    float s = 0.f, d = 0.f;
    for (int c = 0; c < C; ++c) { float v = xv[c]; s += v * as[c]; d += v * ad[c]; }
    alS[t] = s; alD[t] = d;
}

// ---------------- edge softmax helpers ----------------
__device__ inline unsigned f2key(float f) {
    unsigned b = __float_as_uint(f);
    return (b & 0x80000000u) ? ~b : (b | 0x80000000u);
}
__device__ inline float key2f(unsigned k) {
    unsigned b = (k & 0x80000000u) ? (k & 0x7FFFFFFFu) : ~k;
    return __uint_as_float(b);
}
__device__ inline void edge_endpoints(const int* ei, int e, int E, int& src, int& dst) {
    if (e < E) { src = ei[e]; dst = ei[E + e]; }
    else       { src = dst = e - E; }       // PyG self-loops appended
}
__device__ inline float edge_logit(const float* alS, const float* alD,
                                   int src, int dst, int h, int H) {
    float v = alS[src * H + h] + alD[dst * H + h];
    return v > 0.f ? v : v * NEG_SLOPE;
}

__global__ void k_edge_max(const int* __restrict__ ei, int E, int N, int H,
                           const float* __restrict__ alS, const float* __restrict__ alD,
                           unsigned* __restrict__ mkey) {
    int t = blockIdx.x * blockDim.x + threadIdx.x;
    if (t >= (E + N) * H) return;
    int e = t / H, h = t % H, src, dst;
    edge_endpoints(ei, e, E, src, dst);
    atomicMax(&mkey[dst * H + h], f2key(edge_logit(alS, alD, src, dst, h, H)));
}

__global__ void k_edge_expsum(const int* __restrict__ ei, int E, int N, int H,
                              const float* __restrict__ alS, const float* __restrict__ alD,
                              const unsigned* __restrict__ mkey, float* __restrict__ den) {
    int t = blockIdx.x * blockDim.x + threadIdx.x;
    if (t >= (E + N) * H) return;
    int e = t / H, h = t % H, src, dst;
    edge_endpoints(ei, e, E, src, dst);
    float v  = edge_logit(alS, alD, src, dst, h, H);
    float ex = __expf(v - key2f(mkey[dst * H + h]));
    atomicAdd(&den[dst * H + h], ex);
}

// One wave per (edge,head); lane = channel -> coalesced loads of xw[src] and
// coalesced global_atomic_add_f32 bursts into agg[dst].
__global__ void k_edge_aggregate(const int* __restrict__ ei, int E, int N, int H, int C,
                                 const float* __restrict__ alS, const float* __restrict__ alD,
                                 const unsigned* __restrict__ mkey, const float* __restrict__ den,
                                 const float* __restrict__ xw, float* __restrict__ agg) {
    int tid  = blockIdx.x * blockDim.x + threadIdx.x;
    int wid  = tid >> 5;
    int lane = tid & 31;
    if (wid >= (E + N) * H) return;
    int e = wid / H, h = wid % H, src, dst;
    edge_endpoints(ei, e, E, src, dst);
    float v     = edge_logit(alS, alD, src, dst, h, H);
    float ex    = __expf(v - key2f(mkey[dst * H + h]));
    float alpha = ex / (den[dst * H + h] + GAT_EPS);
    const float* xs = &xw[(size_t)src * H * C + h * C];
    float*       od = &agg[(size_t)dst * H * C + h * C];
    for (int c = lane; c < C; c += 32)
        atomicAdd(&od[c], alpha * xs[c]);
}

// ---------------- bias + relu + cvt to f16 (between layers) ----------------
__global__ void k_bias_relu_f16(const float* __restrict__ agg, const float* __restrict__ b,
                                _Float16* __restrict__ out, int N, int F) {
    int t = blockIdx.x * blockDim.x + threadIdx.x;
    if (t >= N * F) return;
    float v = agg[t] + b[t % F];
    out[t] = (_Float16)(v > 0.f ? v : 0.f);
}

// ---------------- layer-2 bias + final 64x2 FC ----------------
__global__ void k_final_fc(const float* __restrict__ agg2, const float* __restrict__ b2,
                           const float* __restrict__ Wfc, const float* __restrict__ bfc,
                           float* __restrict__ out, int N) {
    int n = blockIdx.x * blockDim.x + threadIdx.x;
    if (n >= N) return;
    float a0 = bfc[0], a1 = bfc[1];
    const float* hv = &agg2[(size_t)n * GAT_OUT];
    for (int c = 0; c < GAT_OUT; ++c) {
        float v = hv[c] + b2[c];
        a0 += v * Wfc[c * 2 + 0];
        a1 += v * Wfc[c * 2 + 1];
    }
    out[n * 2 + 0] = a0;
    out[n * 2 + 1] = a1;
}

// ---------------- launcher ----------------
extern "C" void kernel_launch(void* const* d_in, const int* in_sizes, int n_in,
                              void* d_out, int out_size, void* d_ws, size_t ws_size,
                              hipStream_t stream) {
    const float* x     = (const float*)d_in[0];
    const int*   ei    = (const int*)  d_in[1];
    const float* W1    = (const float*)d_in[2];
    const float* aS1   = (const float*)d_in[3];
    const float* aD1   = (const float*)d_in[4];
    const float* b1    = (const float*)d_in[5];
    const float* W2    = (const float*)d_in[6];
    const float* aS2   = (const float*)d_in[7];
    const float* aD2   = (const float*)d_in[8];
    const float* b2    = (const float*)d_in[9];
    const float* Wfc   = (const float*)d_in[10];
    const float* bfc   = (const float*)d_in[11];
    float* out = (float*)d_out;

    const int N  = in_sizes[0] / GAT_IN;    // 50000
    const int E  = in_sizes[1] / 2;         // 800000
    const int ET = E + N;                   // with self-loops

    // workspace carve-out (256B aligned slices)
    size_t off = 0;
    auto carve = [&](size_t bytes) -> void* {
        void* p = (char*)d_ws + off;
        off += (bytes + 255) & ~(size_t)255;
        return p;
    };
    _Float16* xh   = (_Float16*)carve((size_t)N * GAT_IN * 2);
    _Float16* W1p  = (_Float16*)carve((size_t)GAT_IN * GAT_F1 * 2);
    _Float16* W2p  = (_Float16*)carve((size_t)GAT_F1 * GAT_OUT * 2);
    float*    xw1  = (float*)   carve((size_t)N * GAT_F1 * 4);
    float*    alS1 = (float*)   carve((size_t)N * GAT_HEADS * 4);
    float*    alD1 = (float*)   carve((size_t)N * GAT_HEADS * 4);
    unsigned* m1   = (unsigned*)carve((size_t)N * GAT_HEADS * 4);
    float*    den1 = (float*)   carve((size_t)N * GAT_HEADS * 4);
    float*    agg1 = (float*)   carve((size_t)N * GAT_F1 * 4);
    _Float16* h1h  = (_Float16*)carve((size_t)N * GAT_F1 * 2);
    float*    xw2  = (float*)   carve((size_t)N * GAT_OUT * 4);
    float*    alS2 = (float*)   carve((size_t)N * 4);
    float*    alD2 = (float*)   carve((size_t)N * 4);
    unsigned* m2   = (unsigned*)carve((size_t)N * 4);
    float*    den2 = (float*)   carve((size_t)N * 4);
    float*    agg2 = (float*)   carve((size_t)N * GAT_OUT * 4);
    (void)ws_size;

    const int TB = 256;

    // zero-init accumulators (memset 0 == f32 zero and == -inf max-key sentinel)
    hipMemsetAsync(m1,   0, (size_t)N * GAT_HEADS * 4, stream);
    hipMemsetAsync(den1, 0, (size_t)N * GAT_HEADS * 4, stream);
    hipMemsetAsync(agg1, 0, (size_t)N * GAT_F1 * 4, stream);
    hipMemsetAsync(m2,   0, (size_t)N * 4, stream);
    hipMemsetAsync(den2, 0, (size_t)N * 4, stream);
    hipMemsetAsync(agg2, 0, (size_t)N * GAT_OUT * 4, stream);

    // activations -> f16; weights -> packed f16 fragment layout
    k_cvt_f32_to_f16<<<ceil_div(N * GAT_IN, TB), TB, 0, stream>>>(x, xh, N * GAT_IN);
    k_pack_b<<<ceil_div(GAT_IN * GAT_F1, TB), TB, 0, stream>>>(W1, W1p, GAT_IN, GAT_F1);
    k_pack_b<<<ceil_div(GAT_F1 * GAT_OUT, TB), TB, 0, stream>>>(W2, W2p, GAT_F1, GAT_OUT);

    // ---- layer 1 ----
    {
        dim3 g(ceil_div(N, 64), GAT_F1 / 64);
        k_gemm_f16_wmma<<<g, TB, 0, stream>>>(xh, W1p, xw1, N, GAT_IN, GAT_F1);
    }
    k_attn_logits<<<ceil_div(N * GAT_HEADS, TB), TB, 0, stream>>>(
        xw1, aS1, aD1, alS1, alD1, N, GAT_HEADS, GAT_HID);
    {
        int nt = ET * GAT_HEADS;
        k_edge_max<<<ceil_div(nt, TB), TB, 0, stream>>>(ei, E, N, GAT_HEADS, alS1, alD1, m1);
        k_edge_expsum<<<ceil_div(nt, TB), TB, 0, stream>>>(ei, E, N, GAT_HEADS, alS1, alD1, m1, den1);
        k_edge_aggregate<<<ceil_div(nt * 32, TB), TB, 0, stream>>>(
            ei, E, N, GAT_HEADS, GAT_HID, alS1, alD1, m1, den1, xw1, agg1);
    }
    k_bias_relu_f16<<<ceil_div(N * GAT_F1, TB), TB, 0, stream>>>(agg1, b1, h1h, N, GAT_F1);

    // ---- layer 2 (heads=1, ch=64, mean over 1 head == identity) ----
    {
        dim3 g(ceil_div(N, 64), GAT_OUT / 64);
        k_gemm_f16_wmma<<<g, TB, 0, stream>>>(h1h, W2p, xw2, N, GAT_F1, GAT_OUT);
    }
    k_attn_logits<<<ceil_div(N, TB), TB, 0, stream>>>(xw2, aS2, aD2, alS2, alD2, N, 1, GAT_OUT);
    {
        k_edge_max<<<ceil_div(ET, TB), TB, 0, stream>>>(ei, E, N, 1, alS2, alD2, m2);
        k_edge_expsum<<<ceil_div(ET, TB), TB, 0, stream>>>(ei, E, N, 1, alS2, alD2, m2, den2);
        k_edge_aggregate<<<ceil_div(ET * 32, TB), TB, 0, stream>>>(
            ei, E, N, 1, GAT_OUT, alS2, alD2, m2, den2, xw2, agg2);
    }

    // ---- bias + final FC ----
    k_final_fc<<<ceil_div(N, TB), TB, 0, stream>>>(agg2, b2, Wfc, bfc, out, N);
    (void)out_size; (void)n_in;
}
